// SelfAttention_14903536517440
// MI455X (gfx1250) — compile-verified, matrix operand order checked by reading, not run on hardware
//
#include <hip/hip_runtime.h>

#define SEQL   3072
#define DIMN   1536
#define NHEADS 12
#define HDIM   128

typedef __bf16 bf16;
typedef __attribute__((ext_vector_type(16))) __bf16 v16bf;
typedef __attribute__((ext_vector_type(8)))  float  v8f;

struct alignas(16) U128 { unsigned int x[4]; };
struct U256 { U128 lo, hi; };
struct B16x16 { bf16 h[16]; };

typedef __attribute__((address_space(3))) const char* lds_cptr;

// Raw 32-bit LDS byte offset of a __shared__ object (VDST operand of async loads).
__device__ __forceinline__ unsigned lds_off(const void* p) {
  return (unsigned)(unsigned long long)(lds_cptr)p;
}

// CDNA5 async copy: LDS[laddr] = MEM[sbase + voff] (16 bytes, per lane).
// GVS addressing per CDNA5 ISA; tracked by ASYNCcnt.
__device__ __forceinline__ void async_ld_b128(const void* sbase, unsigned voff,
                                              unsigned laddr) {
  asm volatile("global_load_async_to_lds_b128 %0, %1, %2"
               :: "v"(laddr), "v"(voff), "s"(sbase) : "memory");
}
__device__ __forceinline__ void wait_async() {
  asm volatile("s_wait_asynccnt 0" ::: "memory");
}

// Fragment load matching CDNA5 16-bit A/B^T lane layout:
// caller passes pointer already offset by klo; loads halfwords [0..7] and [16..23].
__device__ __forceinline__ v16bf load_frag(const bf16* p) {
  U256 r;
  r.lo = *(const U128*)(p);
  r.hi = *(const U128*)(p + 16);
  return __builtin_bit_cast(v16bf, r);
}

__device__ __forceinline__ float rowmax16(float v) {
  v = fmaxf(v, __shfl_xor(v, 1, 32));
  v = fmaxf(v, __shfl_xor(v, 2, 32));
  v = fmaxf(v, __shfl_xor(v, 4, 32));
  v = fmaxf(v, __shfl_xor(v, 8, 32));
  return v;
}
__device__ __forceinline__ float rowsum16(float v) {
  v += __shfl_xor(v, 1, 32);
  v += __shfl_xor(v, 2, 32);
  v += __shfl_xor(v, 4, 32);
  v += __shfl_xor(v, 8, 32);
  return v;
}

// ---------------------------------------------------------------------------
// One-pass weight prep: Wt[n][k] = bf16(W[k][n]). 64x64 tiles via LDS.
// ---------------------------------------------------------------------------
__global__ __launch_bounds__(256) void convert_wt_kernel(
    const float* __restrict__ W, bf16* __restrict__ Wt, int N, int K)
{
  __shared__ bf16 tile[64 * 65];
  const int tid = threadIdx.x;
  const int k0 = blockIdx.y * 64;
  const int n0 = blockIdx.x * 64;
#pragma unroll
  for (int j = 0; j < 16; ++j) {
    int idx = tid + j * 256;
    int k = idx >> 6, n = idx & 63;
    tile[k * 65 + n] = (bf16)W[(size_t)(k0 + k) * N + n0 + n];
  }
  __syncthreads();
#pragma unroll
  for (int j = 0; j < 16; ++j) {
    int idx = tid + j * 256;
    int n = idx >> 6, k = idx & 63;
    Wt[(size_t)(n0 + n) * K + k0 + k] = tile[k * 65 + n];
  }
}

// ---------------------------------------------------------------------------
// GEMM: C[M,N](bf16) = bf16round( A[M,K](bf16) x Bt[N,K](bf16)^T + bias[N] )
// Block tile 128x64, BK=32, 256 threads = 8 waves, one 16-row strip per wave.
// Double-buffered async staging (global_load_async_to_lds_b128): the DMA for
// K-step i+1 overlaps the WMMA stream of K-step i; one barrier per step.
// ---------------------------------------------------------------------------
__global__ __launch_bounds__(256) void gemm_bf16_kernel(
    const bf16* __restrict__ A, const bf16* __restrict__ Bt,
    const float* __restrict__ bias, bf16* __restrict__ C,
    int M, int N, int K)
{
  const int LDA = 40;  // halfwords, padded (80B rows, 16B aligned)
  const int LDB = 40;
  __shared__ bf16 As[2][128 * 40];
  __shared__ bf16 Bs[2][64 * 40];
  const unsigned ABUF = 128 * 40 * 2;  // bytes per buffer
  const unsigned BBUF = 64 * 40 * 2;

  const int tid  = threadIdx.x;
  const int wave = tid >> 5;
  const int lane = tid & 31;
  const int frow = lane & 15;
  const int klo  = (lane < 16) ? 0 : 8;
  const int row0 = blockIdx.y * 128;
  const int col0 = blockIdx.x * 64;

  // per-thread staging coordinates (constant across K loop)
  const int ar = tid >> 1;              // A row 0..127
  const int ac = (tid & 1) * 16;        // A col (halfwords)
  const unsigned a_lds = lds_off(&As[0][ar * LDA + ac]);
  const int bn = tid >> 2;              // B row (n) 0..63
  const int bc = (tid & 3) * 8;         // B col (halfwords)
  const unsigned b_lds = lds_off(&Bs[0][bn * LDB + bc]);
  const unsigned avoff = (unsigned)(ar * K + ac) * 2u;
  const unsigned bvoff = (unsigned)(bn * K + bc) * 2u;

  auto stage = [&](int buf, int k0) {
    const bf16* abase = A  + (size_t)row0 * K + k0;
    const bf16* bbase = Bt + (size_t)col0 * K + k0;
    unsigned al = a_lds + (unsigned)buf * ABUF;
    async_ld_b128(abase, avoff,      al);
    async_ld_b128(abase, avoff + 16, al + 16);
    async_ld_b128(bbase, bvoff, b_lds + (unsigned)buf * BBUF);
  };

  v8f acc[4] = {};
  stage(0, 0);
  int cur = 0;

  for (int k0 = 0; k0 < K; k0 += 32) {
    wait_async();
    __syncthreads();   // cur fully written by all waves; cur^1 fully consumed
    if (k0 + 32 < K) stage(cur ^ 1, k0 + 32);

    const bf16* Asc = As[cur];
    const bf16* Bsc = Bs[cur];
    v16bf afrag = load_frag(Asc + (wave * 16 + frow) * LDA + klo);
    v16bf bf0 = load_frag(Bsc + (0 * 16 + frow) * LDB + klo);
    v16bf bf1 = load_frag(Bsc + (1 * 16 + frow) * LDB + klo);
    v16bf bf2 = load_frag(Bsc + (2 * 16 + frow) * LDB + klo);
    v16bf bf3 = load_frag(Bsc + (3 * 16 + frow) * LDB + klo);
    acc[0] = __builtin_amdgcn_wmma_f32_16x16x32_bf16(
        false, afrag, false, bf0, (short)0, acc[0], false, false);
    acc[1] = __builtin_amdgcn_wmma_f32_16x16x32_bf16(
        false, afrag, false, bf1, (short)0, acc[1], false, false);
    acc[2] = __builtin_amdgcn_wmma_f32_16x16x32_bf16(
        false, afrag, false, bf2, (short)0, acc[2], false, false);
    acc[3] = __builtin_amdgcn_wmma_f32_16x16x32_bf16(
        false, afrag, false, bf3, (short)0, acc[3], false, false);
    cur ^= 1;
  }

  const int rbase = row0 + wave * 16 + ((lane < 16) ? 0 : 8);
#pragma unroll
  for (int nt = 0; nt < 4; ++nt) {
    int c = col0 + nt * 16 + frow;
    float b = bias[c];
#pragma unroll
    for (int e = 0; e < 8; ++e)
      C[(size_t)(rbase + e) * N + c] = (bf16)(acc[nt][e] + b);
  }
}

// ---------------------------------------------------------------------------
// RMSNorm (over full DIM) + weight + factorized RoPE, in place. 1 block / row.
// ---------------------------------------------------------------------------
__global__ __launch_bounds__(256) void rms_rope_kernel(
    bf16* __restrict__ X, const float* __restrict__ w,
    const float* __restrict__ freqs /* [1024][64][2] */)
{
  const int s   = blockIdx.x;
  const int tid = threadIdx.x;
  bf16* row = X + (size_t)s * DIMN;

  float ss = 0.f;
#pragma unroll
  for (int j = 0; j < 6; ++j) {
    float v = (float)row[tid + j * 256];
    ss += v * v;
  }
  __shared__ float red[256];
  red[tid] = ss;
  __syncthreads();
  for (int off = 128; off > 0; off >>= 1) {
    if (tid < off) red[tid] += red[tid + off];
    __syncthreads();
  }
  const float rinv = rsqrtf(red[0] / (float)DIMN + 1e-6f);

  const int f  = s >> 8;
  const int hh = (s & 255) >> 4;
  const int ww = s & 15;

#pragma unroll
  for (int j = 0; j < 3; ++j) {
    int p = tid + j * 256;        // pair index 0..767
    int i = p & 63;               // within-head pair index (c = 64)
    int e = p * 2;                // even element column
    int pos = (i < 22) ? f : ((i < 43) ? hh : ww);
    const float* fr = freqs + ((size_t)pos * 64 + i) * 2;
    float cs = fr[0], sn = fr[1];
    // emulate reference rounding: bf16(rms) * bf16(w) -> bf16, rope in f32
    float y1 = (float)(bf16)((float)row[e]     * rinv);
    float y2 = (float)(bf16)((float)row[e + 1] * rinv);
    float z1 = (float)(bf16)(y1 * (float)(bf16)w[e]);
    float z2 = (float)(bf16)(y2 * (float)(bf16)w[e + 1]);
    row[e]     = (bf16)(z1 * cs - z2 * sn);
    row[e + 1] = (bf16)(z1 * sn + z2 * cs);
  }
}

// ---------------------------------------------------------------------------
// Flash attention with frame-causal mask. Block = (128 q-rows, head),
// 8 waves each own 16 rows. kv in blocks of 32, double-buffered staging:
// K via async DMA to LDS, V transposed manually; one barrier per kv step.
// ---------------------------------------------------------------------------
__global__ __launch_bounds__(256) void attn_kernel(
    const bf16* __restrict__ Q, const bf16* __restrict__ Kb,
    const bf16* __restrict__ V, bf16* __restrict__ O)
{
  const int LK = 136;   // Ks stride (halfwords): 272B rows, 16B aligned
  const int LV = 40;    // Vt stride
  const int LP = 40;    // P scratch stride
  __shared__ bf16 Ks[2][32 * 136];
  __shared__ bf16 Vt[2][128 * 40];
  __shared__ bf16 Ps[8 * 16 * 40];
  const unsigned KBUF = 32 * 136 * 2;  // bytes per K buffer

  const int h    = blockIdx.y;
  const int qb0  = blockIdx.x * 128;
  const int tid  = threadIdx.x;
  const int wave = tid >> 5;
  const int lane = tid & 31;
  const int frow = lane & 15;
  const int klo  = (lane < 16) ? 0 : 8;

  // Q fragments: 4 chunks of 32 over head_dim, held in VGPRs
  v16bf qf[4];
  {
    const int qrow = qb0 + wave * 16 + frow;
    const bf16* qp = Q + (size_t)qrow * DIMN + h * HDIM;
#pragma unroll
    for (int kc = 0; kc < 4; ++kc)
      qf[kc] = load_frag(qp + kc * 32 + klo);
  }

  v8f oacc[8] = {};
  float m[8], l[8];
#pragma unroll
  for (int e = 0; e < 8; ++e) { m[e] = -1e30f; l[e] = 0.f; }

  const int qrow_base = qb0 + wave * 16 + ((lane < 16) ? 0 : 8);
  const int rframe    = qrow_base >> 8;          // rows base..base+7 share frame
  const int kv_end    = (((qb0 + 127) >> 8) + 1) << 8;
  const float scale   = 0.08838834764831845f;    // 1/sqrt(128)

  // staging coordinates
  const int kr = tid >> 3;              // K row 0..31
  const int kd = (tid & 7) * 16;        // K col (halfwords)
  const unsigned k_lds = lds_off(&Ks[0][kr * LK + kd]);
  const unsigned k_voff = (unsigned)(kr * DIMN + kd) * 2u;
  const int vkv = tid & 31;             // V row (kv)
  const int vd0 = (tid >> 5) * 16;      // V col start

  auto stage = [&](int buf, int kv0) {
    const bf16* kbase = Kb + (size_t)kv0 * DIMN + h * HDIM;
    unsigned kl = k_lds + (unsigned)buf * KBUF;
    async_ld_b128(kbase, k_voff,      kl);
    async_ld_b128(kbase, k_voff + 16, kl + 16);
    const bf16* src = V + (size_t)(kv0 + vkv) * DIMN + h * HDIM + vd0;
    U256 raw;
    raw.lo = *(const U128*)(src);
    raw.hi = *(const U128*)(src + 8);
    B16x16 t = __builtin_bit_cast(B16x16, raw);
    bf16* vt = Vt[buf];
#pragma unroll
    for (int j = 0; j < 16; ++j)
      vt[(vd0 + j) * LV + vkv] = t.h[j];
  };

  stage(0, 0);
  int cur = 0;

  for (int kv0 = 0; kv0 < kv_end; kv0 += 32) {
    wait_async();
    __syncthreads();   // cur populated everywhere; cur^1 fully consumed
    if (kv0 + 32 < kv_end) stage(cur ^ 1, kv0 + 32);

    const bf16* Ksc = Ks[cur];
    const bf16* Vtc = Vt[cur];

    // S = Q K^T for two 16-wide kv tiles
    v8f s0 = {}, s1 = {};
#pragma unroll
    for (int kc = 0; kc < 4; ++kc) {
      v16bf b0 = load_frag(Ksc + frow * LK + kc * 32 + klo);
      s0 = __builtin_amdgcn_wmma_f32_16x16x32_bf16(
          false, qf[kc], false, b0, (short)0, s0, false, false);
      v16bf b1 = load_frag(Ksc + (16 + frow) * LK + kc * 32 + klo);
      s1 = __builtin_amdgcn_wmma_f32_16x16x32_bf16(
          false, qf[kc], false, b1, (short)0, s1, false, false);
    }

    // scale + frame-causal mask
    const bool mask0 = ((kv0 + frow) >> 8)      > rframe;
    const bool mask1 = ((kv0 + 16 + frow) >> 8) > rframe;
#pragma unroll
    for (int e = 0; e < 8; ++e) {
      s0[e] = mask0 ? -1e30f : s0[e] * scale;
      s1[e] = mask1 ? -1e30f : s1[e] * scale;
    }

    // online softmax (row = 16 lanes of a half-wave)
#pragma unroll
    for (int e = 0; e < 8; ++e) {
      float mx   = rowmax16(fmaxf(s0[e], s1[e]));
      float mnew = fmaxf(m[e], mx);
      float alph = __expf(m[e] - mnew);
      float p0   = __expf(s0[e] - mnew);
      float p1   = __expf(s1[e] - mnew);
      l[e] = l[e] * alph + rowsum16(p0 + p1);
      m[e] = mnew;
      s0[e] = p0; s1[e] = p1;
#pragma unroll
      for (int d = 0; d < 8; ++d) oacc[d][e] *= alph;
    }

    // P: C-layout -> A-layout via per-wave LDS scratch
    bf16* pw = Ps + wave * 16 * LP;
#pragma unroll
    for (int e = 0; e < 8; ++e) {
      int r = e + ((lane < 16) ? 0 : 8);
      pw[r * LP + frow]      = (bf16)s0[e];
      pw[r * LP + 16 + frow] = (bf16)s1[e];
    }
    asm volatile("s_wait_dscnt 0" ::: "memory");  // LDS RAW within wave (CDNA5 split counters)
    v16bf pf = load_frag(pw + frow * LP + klo);

    // O += P V
#pragma unroll
    for (int dt = 0; dt < 8; ++dt) {
      v16bf vf = load_frag(Vtc + (dt * 16 + frow) * LV + klo);
      oacc[dt] = __builtin_amdgcn_wmma_f32_16x16x32_bf16(
          false, pf, false, vf, (short)0, oacc[dt], false, false);
    }
    cur ^= 1;
  }

  // epilogue: O /= l, store bf16 into [seq][DIM] with head offset
#pragma unroll
  for (int dt = 0; dt < 8; ++dt) {
    int c = h * HDIM + dt * 16 + frow;
#pragma unroll
    for (int e = 0; e < 8; ++e)
      O[(size_t)(qrow_base + e) * DIMN + c] = (bf16)(oacc[dt][e] / l[e]);
  }
}

// ---------------------------------------------------------------------------
extern "C" void kernel_launch(void* const* d_in, const int* in_sizes, int n_in,
                              void* d_out, int out_size, void* d_ws, size_t ws_size,
                              hipStream_t stream) {
  (void)in_sizes; (void)n_in; (void)out_size; (void)ws_size;
  const bf16*  x     = (const bf16*)d_in[0];
  const float* qw    = (const float*)d_in[1];
  const float* qb    = (const float*)d_in[2];
  const float* kw    = (const float*)d_in[3];
  const float* kb    = (const float*)d_in[4];
  const float* vw    = (const float*)d_in[5];
  const float* vb    = (const float*)d_in[6];
  const float* ow    = (const float*)d_in[7];
  const float* ob    = (const float*)d_in[8];
  const float* nqw   = (const float*)d_in[9];
  const float* nkw   = (const float*)d_in[10];
  const float* freqs = (const float*)d_in[11];
  bf16* out = (bf16*)d_out;

  const size_t elems = (size_t)SEQL * DIMN;
  const size_t welems = (size_t)DIMN * DIMN;
  bf16* q  = (bf16*)d_ws;
  bf16* k  = q + elems;
  bf16* v  = k + elems;
  bf16* o  = v + elems;
  bf16* wqT = o + elems;
  bf16* wkT = wqT + welems;
  bf16* wvT = wkT + welems;
  bf16* woT = wvT + welems;

  dim3 gt(DIMN / 64, DIMN / 64);
  convert_wt_kernel<<<gt, 256, 0, stream>>>(qw, wqT, DIMN, DIMN);
  convert_wt_kernel<<<gt, 256, 0, stream>>>(kw, wkT, DIMN, DIMN);
  convert_wt_kernel<<<gt, 256, 0, stream>>>(vw, wvT, DIMN, DIMN);
  convert_wt_kernel<<<gt, 256, 0, stream>>>(ow, woT, DIMN, DIMN);

  dim3 gg(DIMN / 64, SEQL / 128);
  gemm_bf16_kernel<<<gg, 256, 0, stream>>>(x, wqT, qb, q, SEQL, DIMN, DIMN);
  gemm_bf16_kernel<<<gg, 256, 0, stream>>>(x, wkT, kb, k, SEQL, DIMN, DIMN);
  gemm_bf16_kernel<<<gg, 256, 0, stream>>>(x, wvT, vb, v, SEQL, DIMN, DIMN);
  rms_rope_kernel<<<SEQL, 256, 0, stream>>>(q, nqw, freqs);
  rms_rope_kernel<<<SEQL, 256, 0, stream>>>(k, nkw, freqs);
  attn_kernel<<<dim3(SEQL / 128, NHEADS), 256, 0, stream>>>(q, k, v, o);
  gemm_bf16_kernel<<<gg, 256, 0, stream>>>(o, woT, ob, out, SEQL, DIMN, DIMN);
}
